// MultiHeadAttention_23433341567721
// MI455X (gfx1250) — matrix operand order repaired and checked by reading, not compiled, not run on hardware
//
#include <hip/hip_runtime.h>
#include <hip/hip_bf16.h>

typedef _Float16 h16v __attribute__((ext_vector_type(16)));
typedef _Float16 h8v  __attribute__((ext_vector_type(8)));
typedef float    f8v  __attribute__((ext_vector_type(8)));

#define EMB   512
#define HEADS 8
#define HDIM  64
#define SEQ   2048
#define BATCH 4
#define MTOK  (BATCH * SEQ)   // 8192 tokens

// ---- WMMA helpers -------------------------------------------------------

__device__ __forceinline__ f8v wmma_f16(h16v a, h16v b, f8v c) {
  // D = A(16x32 f16) * B(32x16 f16) + C(16x16 f32)
  return __builtin_amdgcn_wmma_f32_16x16x32_f16(false, a, false, b,
                                                (short)0, c, false, false);
}

// Load an A fragment (16x32, f16) for this lane from a K-contiguous row.
// Lane L row M=L&15 holds K in {base..base+15} (contiguous, matching B)
// with base = (lane>>4)*16; rowptr must already point at element [M][0].
__device__ __forceinline__ h16v load_a16(const _Float16* rowptr, int k0, int lane) {
  int bk = k0 + ((lane >> 4) << 4);
  h8v lo = *(const h8v*)(rowptr + bk);
  h8v hi = *(const h8v*)(rowptr + bk + 8);
  return __builtin_shufflevector(lo, hi, 0, 1, 2, 3, 4, 5, 6, 7,
                                 8, 9, 10, 11, 12, 13, 14, 15);
}

// B fragment (32x16, f16): lane holds column n = lane&15, K = (lane>>4)*16 + i,
// i.e. one contiguous 16-half (32B) run when the source is K-contiguous per column.

// ---- CDNA5 async global->LDS copy (ASYNCcnt) ---------------------------
// Per-lane: copies 16B from the lane's global address directly into LDS at
// the lane's 32-bit LDS offset (low 32 bits of the generic pointer).

__device__ __forceinline__ void async_copy_b128(unsigned lds_off, const void* gptr) {
  asm volatile("global_load_async_to_lds_b128 %0, %1, off"
               :: "v"(lds_off), "v"((unsigned long long)(uintptr_t)gptr)
               : "memory");
}

__device__ __forceinline__ void wait_async0() {
  asm volatile("s_wait_asynccnt 0x0" ::: "memory");
}

// ---- f32 -> f16 conversion ---------------------------------------------

__global__ void __launch_bounds__(256) cvt_f16(const float* __restrict__ src,
                                               _Float16* __restrict__ dst, int n) {
  int i = blockIdx.x * blockDim.x + threadIdx.x;
  int stride = gridDim.x * blockDim.x;
  for (; i < n; i += stride) dst[i] = (_Float16)src[i];
}

// ---- QKV projection: out = x @ W^T + b, stored per-head ----------------
// grid = (MTOK/16/8, EMB/64, 3), block = 256 (8 waves, one 16x64 tile each)

__global__ void __launch_bounds__(256) qkv_gemm(
    const _Float16* __restrict__ xh,
    const _Float16* __restrict__ Wqh, const _Float16* __restrict__ Wkh,
    const _Float16* __restrict__ Wvh,
    const float* __restrict__ bq, const float* __restrict__ bk,
    const float* __restrict__ bv,
    _Float16* __restrict__ Qh, _Float16* __restrict__ Kh,
    _Float16* __restrict__ Vt) {
  const int lane  = threadIdx.x & 31;
  const int wave  = threadIdx.x >> 5;
  const int mtile = blockIdx.x * 8 + wave;   // 0..511
  const int ncol0 = blockIdx.y * 64;
  const int z     = blockIdx.z;              // 0=Q 1=K 2=V

  const _Float16* W   = (z == 0) ? Wqh : (z == 1) ? Wkh : Wvh;
  const float*   bias = (z == 0) ? bq  : (z == 1) ? bk  : bv;

  const _Float16* arow = xh + (size_t)(mtile * 16 + (lane & 15)) * EMB;
  const int nrow = ncol0 + (lane & 15);

  f8v acc[4] = {};
  for (int k0 = 0; k0 < EMB; k0 += 32) {
    h16v a = load_a16(arow, k0, lane);
    const int kb = k0 + ((lane >> 4) << 4);
#pragma unroll
    for (int j = 0; j < 4; ++j) {
      h16v b = *(const h16v*)(W + (size_t)(nrow + j * 16) * EMB + kb);
      acc[j] = wmma_f16(a, b, acc[j]);
    }
  }

  // C/D layout: row m = r + (lane>>4)*8, col n = lane&15 (within 16-wide tile)
  const int rbase = mtile * 16 + ((lane >> 4) << 3);
#pragma unroll
  for (int j = 0; j < 4; ++j) {
    const int col = ncol0 + j * 16 + (lane & 15);
    const float bb = bias[col];
    const int h = col >> 6, d = col & 63;
#pragma unroll
    for (int r = 0; r < 8; ++r) {
      const int m = rbase + r;
      const int bi = m >> 11, t = m & (SEQ - 1);
      const float v = acc[j][r] + bb;
      if (z == 2) {
        Vt[((size_t)(bi * HEADS + h) * HDIM + d) * SEQ + t] = (_Float16)v;
      } else {
        _Float16* dst = (z == 0) ? Qh : Kh;
        dst[((size_t)(bi * HEADS + h) * SEQ + t) * HDIM + d] = (_Float16)v;
      }
    }
  }
}

// ---- Flash attention: softmax(q@k^T)/sqrt(EMB) @ v ---------------------
// grid = (BATCH*HEADS, SEQ/16/8), block = 256; each wave owns 16 queries.
// K/V tiles are shared by all 8 waves: double-buffered in LDS and filled
// with async global->LDS DMA so the next tile's copy overlaps compute.

__global__ void __launch_bounds__(256) attn(
    const _Float16* __restrict__ Qh, const _Float16* __restrict__ Kh,
    const _Float16* __restrict__ Vt, _Float16* __restrict__ ctx) {
  __shared__ _Float16 Kl[2][32][72];  // 32 keys x 64 d (+pad), double buffered
  __shared__ _Float16 Vl[2][64][40];  // 64 d x 32 keys (+pad), double buffered
  __shared__ _Float16 Pl[8][16][40];  // per-wave 16x32 prob tile (+pad)

  const int tid   = threadIdx.x;
  const int lane  = tid & 31;
  const int wave  = tid >> 5;
  const int bh    = blockIdx.x;                 // 0..31
  const int qtile = blockIdx.y * 8 + wave;      // 0..127
  const int q0    = qtile * 16;
  const int hi    = lane >> 4;
  const int nn    = lane & 15;

  const _Float16* Qrow = Qh + ((size_t)bh * SEQ + q0 + nn) * HDIM;
  const h16v aq0 = load_a16(Qrow, 0, lane);
  const h16v aq1 = load_a16(Qrow, 32, lane);

  const _Float16* Kbase = Kh + (size_t)bh * SEQ * HDIM;
  const _Float16* Vbase = Vt + (size_t)bh * HDIM * SEQ;

  // cooperative staging coordinates: each thread copies one 16B chunk per tile
  const int krow = tid >> 3, kc = (tid & 7) * 8;   // K tile: 32 rows x 8 chunks
  const int vrow = tid >> 2, vc = (tid & 3) * 8;   // V tile: 64 rows x 4 chunks
  const _Float16* kgp = Kbase + (size_t)krow * HDIM + kc;
  const _Float16* vgp = Vbase + (size_t)vrow * SEQ + vc;
  const unsigned kl0 = (unsigned)(uintptr_t)&Kl[0][krow][kc];
  const unsigned kl1 = (unsigned)(uintptr_t)&Kl[1][krow][kc];
  const unsigned vl0 = (unsigned)(uintptr_t)&Vl[0][vrow][vc];
  const unsigned vl1 = (unsigned)(uintptr_t)&Vl[1][vrow][vc];

  // preload tile 0 into buffer 0
  async_copy_b128(kl0, kgp);
  async_copy_b128(vl0, vgp);

  f8v O0 = {}, O1 = {}, O2 = {}, O3 = {};
  float mrow[8], lrow[8];
#pragma unroll
  for (int r = 0; r < 8; ++r) { mrow[r] = -__builtin_inff(); lrow[r] = 0.f; }

  int buf = 0;
  for (int kt = 0; kt < SEQ; kt += 32) {
    wait_async0();     // this wave's async copies have landed in LDS
    __syncthreads();   // all waves' copies landed; prev reads of other buf done

    if (kt + 32 < SEQ) {  // start DMA of next tile into the other buffer
      async_copy_b128(buf ? kl0 : kl1, kgp + (size_t)(kt + 32) * HDIM);
      async_copy_b128(buf ? vl0 : vl1, vgp + (kt + 32));
    }

    // scores for 32 keys (two 16-key subtiles), accumulate over d=64
    f8v S0 = {}, S1 = {};
    {
      const _Float16* kr0 = &Kl[buf][nn][hi << 4];
      const _Float16* kr1 = &Kl[buf][16 + nn][hi << 4];
      h16v b00 = *(const h16v*)(kr0);
      h16v b01 = *(const h16v*)(kr0 + 32);
      h16v b10 = *(const h16v*)(kr1);
      h16v b11 = *(const h16v*)(kr1 + 32);
      S0 = wmma_f16(aq0, b00, S0);
      S0 = wmma_f16(aq1, b01, S0);
      S1 = wmma_f16(aq0, b10, S1);
      S1 = wmma_f16(aq1, b11, S1);
    }

    // online softmax per row (rows live in half-wave groups of 16 lanes)
#pragma unroll
    for (int r = 0; r < 8; ++r) {
      float t = fmaxf(S0[r], S1[r]);
#pragma unroll
      for (int off = 1; off < 16; off <<= 1)
        t = fmaxf(t, __shfl_xor(t, off, 32));
      const float mnew  = fmaxf(mrow[r], t);
      const float p0    = __expf(S0[r] - mnew);
      const float p1    = __expf(S1[r] - mnew);
      const float scale = __expf(mrow[r] - mnew);
      float s = p0 + p1;
#pragma unroll
      for (int off = 1; off < 16; off <<= 1)
        s += __shfl_xor(s, off, 32);
      lrow[r] = lrow[r] * scale + s;
      mrow[r] = mnew;
      O0[r] *= scale; O1[r] *= scale; O2[r] *= scale; O3[r] *= scale;
      const int m = r + (hi << 3);
      Pl[wave][m][nn]      = (_Float16)p0;
      Pl[wave][m][16 + nn] = (_Float16)p1;
    }

    // re-layout P (C/D layout) -> A fragment via per-wave LDS (DS in-order)
    h16v pa;
    {
      const _Float16* prow = &Pl[wave][nn][0];
      const int bk = hi << 4;
      h8v lo = *(const h8v*)(prow + bk);
      h8v hc = *(const h8v*)(prow + bk + 8);
      pa = __builtin_shufflevector(lo, hc, 0, 1, 2, 3, 4, 5, 6, 7,
                                   8, 9, 10, 11, 12, 13, 14, 15);
    }

    // O += P(16x32) @ V(32x64): B fragments from the staged V tile
    h16v bv0 = *(const h16v*)&Vl[buf][0 * 16 + nn][hi << 4];
    h16v bv1 = *(const h16v*)&Vl[buf][1 * 16 + nn][hi << 4];
    h16v bv2 = *(const h16v*)&Vl[buf][2 * 16 + nn][hi << 4];
    h16v bv3 = *(const h16v*)&Vl[buf][3 * 16 + nn][hi << 4];
    O0 = wmma_f16(pa, bv0, O0);
    O1 = wmma_f16(pa, bv1, O1);
    O2 = wmma_f16(pa, bv2, O2);
    O3 = wmma_f16(pa, bv3, O3);

    buf ^= 1;
  }

  // epilogue: divide by row-sum and by sqrt(EMB) (reference scales probs)
  const float inv_sqrt = 0.04419417382415922f;  // 1/sqrt(512)
  const int b = bh >> 3, h = bh & 7;
#pragma unroll
  for (int r = 0; r < 8; ++r) {
    const int m = q0 + r + (hi << 3);
    const float invl = inv_sqrt / lrow[r];
    const size_t rowoff = ((size_t)(b * SEQ + m)) * EMB + h * HDIM;
    ctx[rowoff + 0 * 16 + nn] = (_Float16)(O0[r] * invl);
    ctx[rowoff + 1 * 16 + nn] = (_Float16)(O1[r] * invl);
    ctx[rowoff + 2 * 16 + nn] = (_Float16)(O2[r] * invl);
    ctx[rowoff + 3 * 16 + nn] = (_Float16)(O3[r] * invl);
  }
}

// ---- Output projection: out = ctx @ Wp^T + bp (f32 out) ----------------
// grid = (MTOK/16/8, EMB/64), block = 256

__global__ void __launch_bounds__(256) proj_gemm(
    const _Float16* __restrict__ ctx, const _Float16* __restrict__ Wph,
    const float* __restrict__ bp, float* __restrict__ out) {
  const int lane  = threadIdx.x & 31;
  const int wave  = threadIdx.x >> 5;
  const int mtile = blockIdx.x * 8 + wave;
  const int ncol0 = blockIdx.y * 64;

  const _Float16* arow = ctx + (size_t)(mtile * 16 + (lane & 15)) * EMB;
  const int nrow = ncol0 + (lane & 15);

  f8v acc[4] = {};
  for (int k0 = 0; k0 < EMB; k0 += 32) {
    h16v a = load_a16(arow, k0, lane);
    const int kb = k0 + ((lane >> 4) << 4);
#pragma unroll
    for (int j = 0; j < 4; ++j) {
      h16v b = *(const h16v*)(Wph + (size_t)(nrow + j * 16) * EMB + kb);
      acc[j] = wmma_f16(a, b, acc[j]);
    }
  }

  const int rbase = mtile * 16 + ((lane >> 4) << 3);
#pragma unroll
  for (int j = 0; j < 4; ++j) {
    const int col = ncol0 + j * 16 + (lane & 15);
    const float bb = bp[col];
#pragma unroll
    for (int r = 0; r < 8; ++r)
      out[(size_t)(rbase + r) * EMB + col] = acc[j][r] + bb;
  }
}

// ---- Launch ------------------------------------------------------------

extern "C" void kernel_launch(void* const* d_in, const int* in_sizes, int n_in,
                              void* d_out, int out_size, void* d_ws, size_t ws_size,
                              hipStream_t stream) {
  (void)in_sizes; (void)n_in; (void)out_size; (void)ws_size;
  const float* x  = (const float*)d_in[0];
  const float* Wq = (const float*)d_in[1];
  const float* bq = (const float*)d_in[2];
  const float* Wk = (const float*)d_in[3];
  const float* bk = (const float*)d_in[4];
  const float* Wv = (const float*)d_in[5];
  const float* bv = (const float*)d_in[6];
  const float* Wp = (const float*)d_in[7];
  const float* bp = (const float*)d_in[8];
  float* out = (float*)d_out;

  // f16 workspace layout (ctx reuses xh: x is dead after qkv_gemm)
  _Float16* xh  = (_Float16*)d_ws;                       // MTOK*EMB
  _Float16* Wqh = xh  + (size_t)MTOK * EMB;              // EMB*EMB
  _Float16* Wkh = Wqh + (size_t)EMB * EMB;
  _Float16* Wvh = Wkh + (size_t)EMB * EMB;
  _Float16* Wph = Wvh + (size_t)EMB * EMB;
  _Float16* Qh  = Wph + (size_t)EMB * EMB;               // MTOK*EMB
  _Float16* Kh  = Qh  + (size_t)MTOK * EMB;              // MTOK*EMB
  _Float16* Vt  = Kh  + (size_t)MTOK * EMB;              // MTOK*EMB
  _Float16* ctx = xh;                                    // reuse

  cvt_f16<<<1024, 256, 0, stream>>>(x,  xh,  MTOK * EMB);
  cvt_f16<<<256,  256, 0, stream>>>(Wq, Wqh, EMB * EMB);
  cvt_f16<<<256,  256, 0, stream>>>(Wk, Wkh, EMB * EMB);
  cvt_f16<<<256,  256, 0, stream>>>(Wv, Wvh, EMB * EMB);
  cvt_f16<<<256,  256, 0, stream>>>(Wp, Wph, EMB * EMB);

  qkv_gemm<<<dim3(MTOK / 16 / 8, EMB / 64, 3), 256, 0, stream>>>(
      xh, Wqh, Wkh, Wvh, bq, bk, bv, Qh, Kh, Vt);

  attn<<<dim3(BATCH * HEADS, SEQ / 16 / 8), 256, 0, stream>>>(Qh, Kh, Vt, ctx);

  proj_gemm<<<dim3(MTOK / 16 / 8, EMB / 64), 256, 0, stream>>>(ctx, Wph, bp, out);
}
